// MACUnit_22728966930670
// MI455X (gfx1250) — compile-verified, matrix operand order checked by reading, not run on hardware
//
#include <hip/hip_runtime.h>
#include <cstdint>
#include <cstddef>

// ---------------------------------------------------------------------------
// MAC cell (Compositional Attention Networks) on gfx1250.
// Big per-step GEMMs run as bf16 WMMA (v_wmma_f32_16x16x32_bf16, fp32 accum),
// 32x64 tile per wave, register double-buffered across k-chunks.
// ---------------------------------------------------------------------------

typedef __attribute__((ext_vector_type(16))) __bf16 v16bf;
typedef __attribute__((ext_vector_type(8)))  __bf16 v8bf;
typedef __attribute__((ext_vector_type(8)))  float  v8f;

#define DB   512
#define LB   30
#define KBS  196
#define BB   256
#define NSTEP 12
#define MBIG (BB * KBS)   // 50176 = 256*196, multiple of 256

union Frag16 { v16bf v; v8bf h[2]; };

// 16-bit A/B fragment, 16x32 layout: lane half (lane>>4) selects K-subgroups
// {8h..8h+7} and {16+8h..16+8h+7}; two contiguous 16B loads per lane.
__device__ __forceinline__ v16bf load_frag(const __bf16* __restrict__ rowptr,
                                           int k0, int half) {
  Frag16 f;
  f.h[0] = *(const v8bf*)(rowptr + k0 + 8 * half);
  f.h[1] = *(const v8bf*)(rowptr + k0 + 16 + 8 * half);
  return f.v;
}

__device__ __forceinline__ v16bf mul_frag(v16bf a, v16bf s) {
#pragma unroll
  for (int i = 0; i < 16; ++i) a[i] = (__bf16)((float)a[i] * (float)s[i]);
  return a;
}

__device__ __forceinline__ float wave_sum(float s) {
#pragma unroll
  for (int off = 16; off > 0; off >>= 1) s += __shfl_down(s, off);
  return s;
}

#define WMMA_BF16(A_, B_, C_) \
  __builtin_amdgcn_wmma_f32_16x16x32_bf16(false, (A_), false, (B_), (short)0, (C_), false, false)

// ---------------------------------------------------------------------------
// Templated bf16 WMMA GEMM:  C[M,N] = act( A[M,Kd] * Bt[N,Kd]^T + bias + res )
// PROLOG=1: A row is elementwise-multiplied by rowscale[row/rows_per_batch,:]
// ACT: 0=none 1=tanh 2=elu ; RES=1: add bf16 residual [M,N]
// OUTF32: 1 -> fp32 out, 0 -> bf16 out
// Block = 256 threads (8 waves). Each wave: 32(M) x 64(N) tile, 8 WMMAs per
// 32-deep k-chunk, next-chunk fragments loaded before current WMMAs.
// Grid = (N/64, M/256); N fixed at 512 here.
// ---------------------------------------------------------------------------
template<int PROLOG, int ACT, int RES, int OUTF32>
__global__ __launch_bounds__(256) void gemm_wmma(
    const __bf16* __restrict__ A, int lda,
    const __bf16* __restrict__ Bt,
    const float* __restrict__ bias,
    const __bf16* __restrict__ res,
    const __bf16* __restrict__ rowscale, int rows_per_batch,
    void* __restrict__ Cout, int M, int N, int Kd)
{
  const int lane   = threadIdx.x & 31;
  const int wave   = threadIdx.x >> 5;
  const int half   = lane >> 4;
  const int l16    = lane & 15;
  const int m_base = blockIdx.y * 256 + wave * 32;
  const int n_base = blockIdx.x * 64;

  const int mrow0 = m_base + l16;
  const int mrow1 = mrow0 + 16;
  const __bf16* arow0 = A + (size_t)mrow0 * lda;
  const __bf16* arow1 = A + (size_t)mrow1 * lda;
  const __bf16* srow0 = nullptr;
  const __bf16* srow1 = nullptr;
  if (PROLOG) {
    srow0 = rowscale + (size_t)(mrow0 / rows_per_batch) * Kd;
    srow1 = rowscale + (size_t)(mrow1 / rows_per_batch) * Kd;
  }
  const __bf16* brow0 = Bt + (size_t)(n_base + 0 * 16 + l16) * Kd;
  const __bf16* brow1 = Bt + (size_t)(n_base + 1 * 16 + l16) * Kd;
  const __bf16* brow2 = Bt + (size_t)(n_base + 2 * 16 + l16) * Kd;
  const __bf16* brow3 = Bt + (size_t)(n_base + 3 * 16 + l16) * Kd;

  v8f acc[8];
  const v8f zero = {0.f, 0.f, 0.f, 0.f, 0.f, 0.f, 0.f, 0.f};
#pragma unroll
  for (int t = 0; t < 8; ++t) acc[t] = zero;

  // ---- prologue: load k-chunk 0 fragments ----
  v16bf a0 = load_frag(arow0, 0, half);
  v16bf a1 = load_frag(arow1, 0, half);
  v16bf s0 = {}, s1 = {};
  if (PROLOG) { s0 = load_frag(srow0, 0, half); s1 = load_frag(srow1, 0, half); }
  v16bf b0 = load_frag(brow0, 0, half);
  v16bf b1 = load_frag(brow1, 0, half);
  v16bf b2 = load_frag(brow2, 0, half);
  v16bf b3 = load_frag(brow3, 0, half);

  for (int k0 = 0; k0 < Kd; k0 += 32) {
    const int kn = (k0 + 32 < Kd) ? (k0 + 32) : k0;   // clamped (values unused on last iter)
    __builtin_prefetch(arow0 + kn + 32, 0, 1);        // global_prefetch_b8
    __builtin_prefetch(arow1 + kn + 32, 0, 1);
    // ---- issue next-chunk loads (overlap with WMMAs below) ----
    v16bf a0n = load_frag(arow0, kn, half);
    v16bf a1n = load_frag(arow1, kn, half);
    v16bf s0n = {}, s1n = {};
    if (PROLOG) { s0n = load_frag(srow0, kn, half); s1n = load_frag(srow1, kn, half); }
    v16bf b0n = load_frag(brow0, kn, half);
    v16bf b1n = load_frag(brow1, kn, half);
    v16bf b2n = load_frag(brow2, kn, half);
    v16bf b3n = load_frag(brow3, kn, half);

    // ---- compute on current chunk (already resident) ----
    v16bf ua0 = a0, ua1 = a1;
    if (PROLOG) { ua0 = mul_frag(a0, s0); ua1 = mul_frag(a1, s1); }
    acc[0] = WMMA_BF16(ua0, b0, acc[0]);
    acc[1] = WMMA_BF16(ua0, b1, acc[1]);
    acc[2] = WMMA_BF16(ua0, b2, acc[2]);
    acc[3] = WMMA_BF16(ua0, b3, acc[3]);
    acc[4] = WMMA_BF16(ua1, b0, acc[4]);
    acc[5] = WMMA_BF16(ua1, b1, acc[5]);
    acc[6] = WMMA_BF16(ua1, b2, acc[6]);
    acc[7] = WMMA_BF16(ua1, b3, acc[7]);

    a0 = a0n; a1 = a1n;
    if (PROLOG) { s0 = s0n; s1 = s1n; }
    b0 = b0n; b1 = b1n; b2 = b2n; b3 = b3n;
  }

  // C/D layout per 16x16 tile: lane -> n = l16 ; VGPR r -> m = 8*half + r
#pragma unroll
  for (int mt = 0; mt < 2; ++mt) {
#pragma unroll
    for (int t = 0; t < 4; ++t) {
      const int n  = n_base + t * 16 + l16;
      const float bv = bias ? bias[n] : 0.f;
#pragma unroll
      for (int r = 0; r < 8; ++r) {
        const int m = m_base + mt * 16 + 8 * half + r;
        const size_t idx = (size_t)m * N + n;
        float v = acc[mt * 4 + t][r] + bv;
        if (RES) v += (float)res[idx];
        if (ACT == 1)      v = tanhf(v);
        else if (ACT == 2) v = (v > 0.f) ? v : expm1f(v);
        if (OUTF32) ((float*)Cout)[idx]  = v;
        else        ((__bf16*)Cout)[idx] = (__bf16)v;
      }
    }
  }
}

// ---------------------------------------------------------------------------
// Small helper kernels
// ---------------------------------------------------------------------------
__global__ void transpose_bf16_kernel(const float* __restrict__ W,
                                      __bf16* __restrict__ Wt, int In, int Out) {
  int idx = blockIdx.x * 256 + threadIdx.x;    // idx over Wt [Out, In]
  if (idx >= In * Out) return;
  int o = idx / In, i = idx - o * In;
  Wt[idx] = (__bf16)W[(size_t)i * Out + o];
}

__global__ void cvt_bf16_kernel(const float* __restrict__ x,
                                __bf16* __restrict__ y, int n) {
  int i = blockIdx.x * 256 + threadIdx.x;
  if (i < n) y[i] = (__bf16)x[i];
}

// out[r, 0:512] = a[r,:] ; out[r, 512:1024] = b[r,:]   (r in [0,256))
__global__ void pack2_bf16_kernel(const float* __restrict__ a,
                                  const float* __restrict__ b,
                                  __bf16* __restrict__ out) {
  int idx = blockIdx.x * 256 + threadIdx.x;    // 256*512 threads
  int r = idx >> 9, d = idx & 511;
  out[(size_t)r * 1024 + d]       = (__bf16)a[idx];
  out[(size_t)r * 1024 + 512 + d] = (__bf16)b[idx];
}

__global__ void init_cm_kernel(const float* __restrict__ ic,
                               const float* __restrict__ im,
                               float* __restrict__ ctrl,
                               float* __restrict__ mem) {
  int idx = blockIdx.x * 256 + threadIdx.x;    // B*D
  int d = idx & 511;
  ctrl[idx] = ic[d];
  mem[idx]  = im[d];
}

__global__ void copy_f32_kernel(const float* __restrict__ x,
                                float* __restrict__ y, int n) {
  int i = blockIdx.x * 256 + threadIdx.x;
  if (i < n) y[i] = x[i];
}

// ControlUnit attention: logits[b,l] = sum_d q[b,d]*cattn_w[d]*ctx[b,l,d] + b0
// softmax over L, ctrl_out[b,:] = sum_l attn*ctx[b,l,:]
__global__ __launch_bounds__(256) void ctrl_attn_kernel(
    const float* __restrict__ q, const float* __restrict__ context,
    const float* __restrict__ cattn_w, const float* __restrict__ cattn_b,
    float* __restrict__ ctrl_out)
{
  const int b = blockIdx.x;
  __shared__ float sh_qc[DB];
  __shared__ float sh_logit[32];
  const int tid = threadIdx.x;
  for (int d = tid; d < DB; d += 256) sh_qc[d] = q[b * DB + d] * cattn_w[d];
  __syncthreads();
  const int lane = tid & 31, wave = tid >> 5;
  const float* ctx = context + (size_t)b * LB * DB;
  for (int l = wave; l < LB; l += 8) {
    float s = 0.f;
    for (int d = lane; d < DB; d += 32) s += ctx[l * DB + d] * sh_qc[d];
    s = wave_sum(s);
    if (lane == 0) sh_logit[l] = s + cattn_b[0];
  }
  __syncthreads();
  if (tid == 0) {
    float mx = sh_logit[0];
    for (int l = 1; l < LB; ++l) mx = fmaxf(mx, sh_logit[l]);
    float sum = 0.f;
    for (int l = 0; l < LB; ++l) { float e = expf(sh_logit[l] - mx); sh_logit[l] = e; sum += e; }
    float inv = 1.f / sum;
    for (int l = 0; l < LB; ++l) sh_logit[l] *= inv;
  }
  __syncthreads();
  for (int d = tid; d < DB; d += 256) {
    float s = 0.f;
    for (int l = 0; l < LB; ++l) s += sh_logit[l] * ctx[l * DB + d];
    ctrl_out[b * DB + d] = s;
  }
}

// ReadUnit attention: logit[b,k] = rb + sum_d elu(Y[b,k,d]*ctrl[b,d])*rattn_w[d]
// softmax over K, read[b,:] = sum_k attn*knowledge[b,k,:]
__global__ __launch_bounds__(256) void read_attn_kernel(
    const __bf16* __restrict__ Y, const float* __restrict__ ctrl,
    const float* __restrict__ rattn_w, const float* __restrict__ rattn_b,
    const float* __restrict__ knowledge, float* __restrict__ read_out)
{
  const int b = blockIdx.x;
  __shared__ float sh_c[DB];
  __shared__ float sh_w[DB];
  __shared__ float sh_logit[KBS];
  const int tid = threadIdx.x;
  for (int d = tid; d < DB; d += 256) {
    sh_c[d] = ctrl[b * DB + d];
    sh_w[d] = rattn_w[d];
  }
  __syncthreads();
  const int lane = tid & 31, wave = tid >> 5;
  const __bf16* Yb = Y + (size_t)b * KBS * DB;
  for (int k = wave; k < KBS; k += 8) {
    float s = 0.f;
    for (int d = lane; d < DB; d += 32) {
      float v = (float)Yb[(size_t)k * DB + d] * sh_c[d];
      v = (v > 0.f) ? v : expm1f(v);
      s += v * sh_w[d];
    }
    s = wave_sum(s);
    if (lane == 0) sh_logit[k] = s + rattn_b[0];
  }
  __syncthreads();
  if (tid == 0) {
    float mx = sh_logit[0];
    for (int k = 1; k < KBS; ++k) mx = fmaxf(mx, sh_logit[k]);
    float sum = 0.f;
    for (int k = 0; k < KBS; ++k) { float e = expf(sh_logit[k] - mx); sh_logit[k] = e; sum += e; }
    float inv = 1.f / sum;
    for (int k = 0; k < KBS; ++k) sh_logit[k] *= inv;
  }
  __syncthreads();
  const float* kb = knowledge + (size_t)b * KBS * DB;
  for (int d = tid; d < DB; d += 256) {
    float s = 0.f;
    for (int k = 0; k < KBS; ++k) s += sh_logit[k] * kb[(size_t)k * DB + d];
    read_out[b * DB + d] = s;
  }
}

// ---------------------------------------------------------------------------
extern "C" void kernel_launch(void* const* d_in, const int* in_sizes, int n_in,
                              void* d_out, int out_size, void* d_ws, size_t ws_size,
                              hipStream_t stream) {
  const float* context    = (const float*)d_in[0];
  const float* knowledge  = (const float*)d_in[1];
  const float* init_ctrl  = (const float*)d_in[2];
  const float* init_mem   = (const float*)d_in[3];
  const float* ci_w       = (const float*)d_in[4];
  const float* ci_b       = (const float*)d_in[5];
  const float* cu_w       = (const float*)d_in[6];
  const float* cu_b       = (const float*)d_in[7];
  const float* cattn_w    = (const float*)d_in[8];
  const float* cattn_b    = (const float*)d_in[9];
  const float* kproj_w    = (const float*)d_in[10];
  const float* kproj_b    = (const float*)d_in[11];
  const float* mproj_w    = (const float*)d_in[12];
  const float* mproj_b    = (const float*)d_in[13];
  const float* concat_w   = (const float*)d_in[14];
  const float* concat_b   = (const float*)d_in[15];
  const float* concat2_w  = (const float*)d_in[16];
  const float* concat2_b  = (const float*)d_in[17];
  const float* rattn_w    = (const float*)d_in[18];
  const float* rattn_b    = (const float*)d_in[19];
  const float* write_w    = (const float*)d_in[20];
  const float* write_b    = (const float*)d_in[21];
  float* out = (float*)d_out;

  // -------- workspace layout --------
  char* p = (char*)d_ws;
  auto alloc = [&](size_t bytes) -> void* {
    void* r = (void*)p;
    p += (bytes + 255) & ~(size_t)255;
    return r;
  };
  __bf16* kprojT   = (__bf16*)alloc((size_t)DB * DB * 2);
  __bf16* W1T      = (__bf16*)alloc((size_t)DB * DB * 2);
  __bf16* W2T      = (__bf16*)alloc((size_t)DB * DB * 2);
  __bf16* concat2T = (__bf16*)alloc((size_t)DB * DB * 2);
  __bf16* mprojT   = (__bf16*)alloc((size_t)DB * DB * 2);
  __bf16* ciT      = (__bf16*)alloc((size_t)DB * 2 * DB * 2);
  __bf16* writeT   = (__bf16*)alloc((size_t)DB * 2 * DB * 2);
  __bf16* wuT      = (__bf16*)alloc((size_t)NSTEP * DB * DB * 2);
  __bf16* kn_bf    = (__bf16*)alloc((size_t)MBIG * DB * 2);
  __bf16* kp_bf    = (__bf16*)alloc((size_t)MBIG * DB * 2);
  __bf16* kp2_bf   = (__bf16*)alloc((size_t)MBIG * DB * 2);
  __bf16* inter_bf = (__bf16*)alloc((size_t)MBIG * DB * 2);
  __bf16* Y_bf     = (__bf16*)alloc((size_t)MBIG * DB * 2);
  __bf16* cm_bf    = (__bf16*)alloc((size_t)BB * 2 * DB * 2);
  __bf16* q1_bf    = (__bf16*)alloc((size_t)BB * DB * 2);
  float*  q_f      = (float*)alloc((size_t)BB * DB * 4);
  __bf16* mp_bf    = (__bf16*)alloc((size_t)BB * DB * 2);
  float*  ctrl_f   = (float*)alloc((size_t)BB * DB * 4);
  float*  mem_f    = (float*)alloc((size_t)BB * DB * 4);
  float*  read_f   = (float*)alloc((size_t)BB * DB * 4);
  __bf16* wr_bf    = (__bf16*)alloc((size_t)BB * 2 * DB * 2);

  const dim3 blk(256);
  auto ggrid = [](int M) { return dim3(DB / 64, M / 256); };

  // -------- one-time prep --------
  {
    int n = DB * DB, b = (n + 255) / 256;
    transpose_bf16_kernel<<<b, blk, 0, stream>>>(kproj_w, kprojT, DB, DB);
    transpose_bf16_kernel<<<b, blk, 0, stream>>>(concat_w, W1T, DB, DB);             // rows 0..D-1
    transpose_bf16_kernel<<<b, blk, 0, stream>>>(concat_w + DB * DB, W2T, DB, DB);   // rows D..2D-1
    transpose_bf16_kernel<<<b, blk, 0, stream>>>(concat2_w, concat2T, DB, DB);
    transpose_bf16_kernel<<<b, blk, 0, stream>>>(mproj_w, mprojT, DB, DB);
    int n2 = 2 * DB * DB, b2 = (n2 + 255) / 256;
    transpose_bf16_kernel<<<b2, blk, 0, stream>>>(ci_w, ciT, 2 * DB, DB);
    transpose_bf16_kernel<<<b2, blk, 0, stream>>>(write_w, writeT, 2 * DB, DB);
    for (int s = 0; s < NSTEP; ++s)
      transpose_bf16_kernel<<<b, blk, 0, stream>>>(cu_w + (size_t)s * DB * DB,
                                                   wuT + (size_t)s * DB * DB, DB, DB);
  }
  {
    int n = MBIG * DB;
    cvt_bf16_kernel<<<(n + 255) / 256, blk, 0, stream>>>(knowledge, kn_bf, n);
  }
  // know_proj = knowledge @ kproj_w + kproj_b    (bf16 out)
  gemm_wmma<0, 0, 0, 0><<<ggrid(MBIG), blk, 0, stream>>>(
      kn_bf, DB, kprojT, kproj_b, nullptr, nullptr, 1, kp_bf, MBIG, DB, DB);
  // kp2 = know_proj @ W2 + concat_b              (hoisted, bf16 out)
  gemm_wmma<0, 0, 0, 0><<<ggrid(MBIG), blk, 0, stream>>>(
      kp_bf, DB, W2T, concat_b, nullptr, nullptr, 1, kp2_bf, MBIG, DB, DB);

  init_cm_kernel<<<(BB * DB) / 256, blk, 0, stream>>>(init_ctrl, init_mem, ctrl_f, mem_f);

  // -------- recurrent steps --------
  for (int s = 0; s < NSTEP; ++s) {
    // pack [control | memory] as bf16
    pack2_bf16_kernel<<<(BB * DB) / 256, blk, 0, stream>>>(ctrl_f, mem_f, cm_bf);
    // q1 = tanh(cm @ ci_w + ci_b)
    gemm_wmma<0, 1, 0, 0><<<ggrid(BB), blk, 0, stream>>>(
        cm_bf, 2 * DB, ciT, ci_b, nullptr, nullptr, 1, q1_bf, BB, DB, 2 * DB);
    // q = q1 @ cu_w[s] + cu_b[s]   (fp32 out for attention)
    gemm_wmma<0, 0, 0, 1><<<ggrid(BB), blk, 0, stream>>>(
        q1_bf, DB, wuT + (size_t)s * DB * DB, cu_b + (size_t)s * DB,
        nullptr, nullptr, 1, q_f, BB, DB, DB);
    // control attention -> new control
    ctrl_attn_kernel<<<BB, blk, 0, stream>>>(q_f, context, cattn_w, cattn_b, ctrl_f);
    // mem_proj = memory @ mproj_w + mproj_b   (bf16 row-scale)
    gemm_wmma<0, 0, 0, 0><<<ggrid(BB), blk, 0, stream>>>(
        cm_bf + DB, 2 * DB, mprojT, mproj_b, nullptr, nullptr, 1, mp_bf, BB, DB, DB);
    // inter = elu( (know_proj . mem_proj) @ W1 + kp2 )
    gemm_wmma<1, 2, 1, 0><<<ggrid(MBIG), blk, 0, stream>>>(
        kp_bf, DB, W1T, nullptr, kp2_bf, mp_bf, KBS, inter_bf, MBIG, DB, DB);
    // Y = inter @ concat2_w + concat2_b
    gemm_wmma<0, 0, 0, 0><<<ggrid(MBIG), blk, 0, stream>>>(
        inter_bf, DB, concat2T, concat2_b, nullptr, nullptr, 1, Y_bf, MBIG, DB, DB);
    // read attention (elu(Y*control) -> softmax_K -> weighted knowledge sum)
    read_attn_kernel<<<BB, blk, 0, stream>>>(Y_bf, ctrl_f, rattn_w, rattn_b,
                                             knowledge, read_f);
    // memory = [memory | read] @ write_w + write_b
    pack2_bf16_kernel<<<(BB * DB) / 256, blk, 0, stream>>>(mem_f, read_f, wr_bf);
    gemm_wmma<0, 0, 0, 1><<<ggrid(BB), blk, 0, stream>>>(
        wr_bf, 2 * DB, writeT, write_b, nullptr, nullptr, 1, mem_f, BB, DB, 2 * DB);
  }

  copy_f32_kernel<<<(BB * DB + 255) / 256, blk, 0, stream>>>(mem_f, out, BB * DB);
}
